// GCN_27462020891063
// MI455X (gfx1250) — matrix operand order repaired and checked by reading, not compiled, not run on hardware
//
#include <hip/hip_runtime.h>
#include <cstdint>
#include <cstddef>

// ---- problem constants (match reference) ----
#define NN      50000
#define NE      800000
#define IND     128
#define H1D     512
#define H2D     300
#define H2P     320      // H2 padded to multiple of 64 (4 WMMA N-tiles per wave)
#define OUTD    10
#define BN_EPS  1e-5f

typedef __attribute__((ext_vector_type(16))) _Float16 v16h;
typedef __attribute__((ext_vector_type(8)))  _Float16 v8h;
typedef __attribute__((ext_vector_type(8)))  float    v8f;

static __host__ int cdiv(long long a, long long b) { return (int)((a + b - 1) / b); }

// -------------------------------------------------------------------------
// small utility kernels
// -------------------------------------------------------------------------
__global__ void k_zero_f32(float* p, int n) {
    int i = blockIdx.x * blockDim.x + threadIdx.x;
    if (i < n) p[i] = 0.0f;
}

__global__ void k_init_deg(float* deg, int n) {
    int i = blockIdx.x * blockDim.x + threadIdx.x;
    if (i < n) deg[i] = 1.0f;           // self-loop weight 1.0
}

__global__ void k_deg_accum(const long long* __restrict__ col,
                            const float* __restrict__ w,
                            float* __restrict__ deg, int e) {
    int i = blockIdx.x * blockDim.x + threadIdx.x;
    if (i < e) atomicAdd(&deg[col[i]], w[i]);
}

__global__ void k_dinv(float* deg, int n) {
    int i = blockIdx.x * blockDim.x + threadIdx.x;
    if (i < n) {
        float d = deg[i];
        deg[i] = (d > 0.0f) ? rsqrtf(d) : 0.0f;   // in place: deg -> dinv
    }
}

__global__ void k_f32_to_f16(const float* __restrict__ in, _Float16* __restrict__ out,
                             long long n) {
    long long i = (long long)blockIdx.x * blockDim.x + threadIdx.x;
    if (i < n) out[i] = (_Float16)in[i];
}

// W [K x Nw] f32 (row-major) -> WT [Npad x K] f16, zero-padded rows
__global__ void k_transpose_f16(const float* __restrict__ W, _Float16* __restrict__ WT,
                                int K, int Nw, int Npad) {
    int idx = blockIdx.x * blockDim.x + threadIdx.x;
    if (idx >= Npad * K) return;
    int n = idx / K;
    int k = idx - n * K;
    WT[idx] = (n < Nw) ? (_Float16)W[(size_t)k * Nw + n] : (_Float16)0.0f;
}

// -------------------------------------------------------------------------
// GCN propagation: self-loop init + edge scatter-add
// -------------------------------------------------------------------------
__global__ void k_agg1_init(const float* __restrict__ x, const float* __restrict__ dinv,
                            float* __restrict__ agg) {
    int idx = blockIdx.x * blockDim.x + threadIdx.x;
    if (idx >= NN * IND) return;
    int r = idx >> 7;                       // / IND
    float di = dinv[r];
    agg[idx] = di * di * x[idx];
}

__global__ void k_agg2_init(const float* __restrict__ t2, const float* __restrict__ dinv,
                            const float* __restrict__ b2, float* __restrict__ agg) {
    int idx = blockIdx.x * blockDim.x + threadIdx.x;
    if (idx >= NN * H2D) return;
    int r = idx / H2D;
    int j = idx - r * H2D;
    float di = dinv[r];
    agg[idx] = di * di * t2[idx] + b2[j];
}

// one thread handles 4 contiguous dims of one edge; G = D/4 groups per edge
__global__ void k_edge_scatter(const long long* __restrict__ rows,
                               const long long* __restrict__ cols,
                               const float* __restrict__ w,
                               const float* __restrict__ dinv,
                               const float* __restrict__ src,
                               float* __restrict__ dst,
                               int D, int G) {
    long long tid = (long long)blockIdx.x * blockDim.x + threadIdx.x;
    long long e = tid / G;
    int g = (int)(tid - e * (long long)G);
    if (e >= NE) return;
    long long r = rows[e], c = cols[e];
    float nrm = dinv[r] * w[e] * dinv[c];
    float4 v = *((const float4*)(src + r * (long long)D) + g);
    float* o = dst + c * (long long)D + 4 * g;
    atomicAdd(o + 0, nrm * v.x);
    atomicAdd(o + 1, nrm * v.y);
    atomicAdd(o + 2, nrm * v.z);
    atomicAdd(o + 3, nrm * v.w);
}

// -------------------------------------------------------------------------
// WMMA GEMM, register-blocked 1Mx4N:
//   C[M x ldc] = A16[M x K] * BT16[Npad x K]^T (+bias), f32 accumulate.
// One wave32 owns a 16x64 strip (4 x 16x16 tiles). Per K-step of 32 it loads
// ONE A fragment + FOUR B fragments (all into distinct live ranges, so loads
// stay in flight), then issues 4 back-to-back WMMAs sharing the A registers.
// Npad must be a multiple of 64; M a multiple of 16.
// -------------------------------------------------------------------------
__global__ void k_wmma_gemm4(const _Float16* __restrict__ A,
                             const _Float16* __restrict__ BT,
                             float* __restrict__ C,
                             const float* __restrict__ bias,
                             int K, int nGrpN, int Nreal, int ldc, int totalGrps) {
    int wavesPerBlk = blockDim.x >> 5;
    int grp = blockIdx.x * wavesPerBlk + (threadIdx.x >> 5);
    if (grp >= totalGrps) return;                  // wave-uniform exit
    int lane = threadIdx.x & 31;
    int hi   = lane >> 4;                          // 0: lanes 0-15, 1: lanes 16-31
    int l15  = lane & 15;

    int tm = grp / nGrpN;
    int gn = grp - tm * nGrpN;                     // group of 4 N-tiles

    long long arow = (long long)(tm * 16 + l15) * K;           // A row (M = lane&15)
    long long brow = (long long)(gn * 64 + l15) * K;           // BT row of tile t=0
    const long long bstep = (long long)16 * K;                 // next N-tile

    v8f acc[4];
#pragma unroll
    for (int t = 0; t < 4; ++t) acc[t] = (v8f){};

#pragma unroll 2
    for (int k0 = 0; k0 < K; k0 += 32) {
        // ---- load phase: 10 independent b128 loads, all in flight ----
        // A fragment: halves [0..7] -> K = k0+8*hi.. ; [8..15] -> K = k0+16+8*hi..
        v8h alo = *(const v8h*)(A + arow + k0 + 8 * hi);
        v8h ahi = *(const v8h*)(A + arow + k0 + 16 + 8 * hi);
        v16h bfr[4];
#pragma unroll
        for (int t = 0; t < 4; ++t) {
            // B fragment: halves [0..7] -> K = k0+8*hi.. ; [8..15] -> K = k0+16+8*hi..
            const _Float16* bp = BT + brow + bstep * t + k0 + 8 * hi;
            v8h blo = *(const v8h*)(bp);
            v8h bhi = *(const v8h*)(bp + 16);
            bfr[t] = __builtin_shufflevector(blo, bhi,
                                             0,1,2,3,4,5,6,7,8,9,10,11,12,13,14,15);
        }
        v16h a = __builtin_shufflevector(alo, ahi,
                                         0,1,2,3,4,5,6,7,8,9,10,11,12,13,14,15);
        // ---- compute phase: 4 WMMAs sharing the A fragment ----
#pragma unroll
        for (int t = 0; t < 4; ++t) {
            acc[t] = __builtin_amdgcn_wmma_f32_16x16x32_f16(false, a, false, bfr[t],
                                                            (short)0, acc[t], false, false);
        }
    }

    // C layout per tile: VGPR r -> row = 16*tm + r + 8*hi, col = tileN*16 + (lane&15)
    int rbase = tm * 16 + hi * 8;
#pragma unroll
    for (int t = 0; t < 4; ++t) {
        int col = gn * 64 + t * 16 + l15;
        if (col < Nreal) {
            float bv = bias ? bias[col] : 0.0f;
#pragma unroll
            for (int r = 0; r < 8; ++r) {
                C[(size_t)(rbase + r) * ldc + col] = acc[t][r] + bv;
            }
        }
    }
}

// -------------------------------------------------------------------------
// BatchNorm: two-pass (sum, sumsq) with LDS tree reduction
// -------------------------------------------------------------------------
__global__ void k_bn_stats(const float* __restrict__ h, int M, int D,
                           float* __restrict__ sums, float* __restrict__ sumsq) {
    __shared__ float ss[256];
    __shared__ float sq[256];
    int colj = blockIdx.x;                 // one column per blockIdx.x
    int tid  = threadIdx.x;
    float s = 0.0f, q = 0.0f;
    for (int r = blockIdx.y * blockDim.x + tid; r < M; r += gridDim.y * blockDim.x) {
        float v = h[(size_t)r * D + colj];
        s += v; q += v * v;
    }
    ss[tid] = s; sq[tid] = q;
    __syncthreads();
    for (int o = blockDim.x >> 1; o > 0; o >>= 1) {
        if (tid < o) { ss[tid] += ss[tid + o]; sq[tid] += sq[tid + o]; }
        __syncthreads();
    }
    if (tid == 0) {
        atomicAdd(&sums[colj],  ss[0]);
        atomicAdd(&sumsq[colj], sq[0]);
    }
}

__global__ void k_bn_finalize(const float* __restrict__ sums, const float* __restrict__ sumsq,
                              const float* __restrict__ g, const float* __restrict__ beta,
                              float* __restrict__ scale, float* __restrict__ shift,
                              int M, int D) {
    int j = blockIdx.x * blockDim.x + threadIdx.x;
    if (j >= D) return;
    float inv_m = 1.0f / (float)M;
    float mean = sums[j] * inv_m;
    float var  = sumsq[j] * inv_m - mean * mean;     // biased variance (torch BN)
    float sc   = g[j] * rsqrtf(var + BN_EPS);
    scale[j] = sc;
    shift[j] = beta[j] - mean * sc;
}

// fused BN + ReLU -> f16 (feeds next WMMA GEMM)
__global__ void k_bn_relu_f16(const float* __restrict__ in,
                              const float* __restrict__ scale, const float* __restrict__ shift,
                              _Float16* __restrict__ out, long long n, int D) {
    long long idx = (long long)blockIdx.x * blockDim.x + threadIdx.x;
    if (idx >= n) return;
    int j = (int)(idx % D);
    out[idx] = (_Float16)fmaxf(0.0f, in[idx] * scale[j] + shift[j]);
}

// fused BN + ReLU in-place f32 (feeds scalar head)
__global__ void k_bn_relu_f32(float* __restrict__ h,
                              const float* __restrict__ scale, const float* __restrict__ shift,
                              long long n, int D) {
    long long idx = (long long)blockIdx.x * blockDim.x + threadIdx.x;
    if (idx >= n) return;
    int j = (int)(idx % D);
    h[idx] = fmaxf(0.0f, h[idx] * scale[j] + shift[j]);
}

// -------------------------------------------------------------------------
// classifier head: relu(h @ Wl + bl) then log_softmax; one thread per node
// -------------------------------------------------------------------------
__global__ void k_head(const float* __restrict__ h, const float* __restrict__ Wl,
                       const float* __restrict__ bl, float* __restrict__ out, int n) {
    int i = blockIdx.x * blockDim.x + threadIdx.x;
    if (i >= n) return;
    float acc[OUTD];
#pragma unroll
    for (int c = 0; c < OUTD; ++c) acc[c] = bl[c];
    const float* hr = h + (size_t)i * H2D;
    for (int k = 0; k < H2D; ++k) {
        float v = hr[k];
        const float* wr = Wl + k * OUTD;
#pragma unroll
        for (int c = 0; c < OUTD; ++c) acc[c] += v * wr[c];
    }
    float m = 0.0f;                                  // relu => max >= 0
#pragma unroll
    for (int c = 0; c < OUTD; ++c) { acc[c] = fmaxf(acc[c], 0.0f); m = fmaxf(m, acc[c]); }
    float s = 0.0f;
#pragma unroll
    for (int c = 0; c < OUTD; ++c) s += expf(acc[c] - m);
    float lse = m + logf(s);
#pragma unroll
    for (int c = 0; c < OUTD; ++c) out[(size_t)i * OUTD + c] = acc[c] - lse;
}

// -------------------------------------------------------------------------
// launcher
// -------------------------------------------------------------------------
extern "C" void kernel_launch(void* const* d_in, const int* in_sizes, int n_in,
                              void* d_out, int out_size, void* d_ws, size_t ws_size,
                              hipStream_t stream) {
    (void)in_sizes; (void)n_in; (void)out_size; (void)ws_size;

    const float*     x     = (const float*)d_in[0];
    const long long* ei    = (const long long*)d_in[1];  // int64 in reference
    const long long* erow  = ei;            // sources
    const long long* ecol  = ei + NE;       // targets
    const float*     ew    = (const float*)d_in[2];
    const float*     W1    = (const float*)d_in[3];
    const float*     b1    = (const float*)d_in[4];
    const float*     g1    = (const float*)d_in[5];
    const float*     beta1 = (const float*)d_in[6];
    const float*     W2    = (const float*)d_in[7];
    const float*     b2    = (const float*)d_in[8];
    const float*     g2    = (const float*)d_in[9];
    const float*     beta2 = (const float*)d_in[10];
    const float*     Wl    = (const float*)d_in[11];
    const float*     bl    = (const float*)d_in[12];
    float*           outp  = (float*)d_out;

    // ---- carve workspace (deterministic layout) ----
    char* p = (char*)d_ws;
    auto carve = [&](size_t bytes) -> void* {
        void* q = (void*)p;
        p += (bytes + 255) & ~(size_t)255;
        return q;
    };
    float*    dinv  = (float*)   carve(sizeof(float)    * NN);
    float*    agg1  = (float*)   carve(sizeof(float)    * (size_t)NN * IND);
    _Float16* a1h   = (_Float16*)carve(sizeof(_Float16) * (size_t)NN * IND);
    _Float16* w1t   = (_Float16*)carve(sizeof(_Float16) * (size_t)H1D * IND);
    float*    y1    = (float*)   carve(sizeof(float)    * (size_t)NN * H1D);
    _Float16* h1h   = (_Float16*)carve(sizeof(_Float16) * (size_t)NN * H1D);
    _Float16* w2t   = (_Float16*)carve(sizeof(_Float16) * (size_t)H2P * H1D);
    float*    t2    = (float*)   carve(sizeof(float)    * (size_t)NN * H2D);
    float*    agg2  = (float*)   carve(sizeof(float)    * (size_t)NN * H2D);
    float*    sums  = (float*)   carve(sizeof(float)    * H1D);
    float*    sumsq = (float*)   carve(sizeof(float)    * H1D);
    float*    scale = (float*)   carve(sizeof(float)    * H1D);
    float*    shift = (float*)   carve(sizeof(float)    * H1D);

    const int B = 256;

    // ---- gcn_norm ----
    k_init_deg <<<cdiv(NN, B), B, 0, stream>>>(dinv, NN);
    k_deg_accum<<<cdiv(NE, B), B, 0, stream>>>(ecol, ew, dinv, NE);
    k_dinv     <<<cdiv(NN, B), B, 0, stream>>>(dinv, NN);

    // ---- layer 1: propagate(x) ----
    k_agg1_init<<<cdiv((long long)NN * IND, B), B, 0, stream>>>(x, dinv, agg1);
    k_edge_scatter<<<cdiv((long long)NE * (IND / 4), B), B, 0, stream>>>(
        erow, ecol, ew, dinv, x, agg1, IND, IND / 4);

    // ---- GEMM1: y1 = agg1 @ W1 + b1 (f16 WMMA, f32 accum, 1x4 blocked) ----
    k_f32_to_f16<<<cdiv((long long)NN * IND, B), B, 0, stream>>>(agg1, a1h, (long long)NN * IND);
    k_transpose_f16<<<cdiv((long long)H1D * IND, B), B, 0, stream>>>(W1, w1t, IND, H1D, H1D);
    {
        int nGrpN = H1D / 64;                          // 8 groups of 4 N-tiles
        int totalGrps = (NN / 16) * nGrpN;             // 3125 * 8
        k_wmma_gemm4<<<cdiv(totalGrps, B / 32), B, 0, stream>>>(
            a1h, w1t, y1, b1, IND, nGrpN, H1D, H1D, totalGrps);
    }

    // ---- BN1 + ReLU -> h1 (f16) ----
    k_zero_f32<<<cdiv(2 * H1D, B), B, 0, stream>>>(sums, 2 * H1D);   // sums+sumsq contiguous
    {
        dim3 g(H1D, 32);
        k_bn_stats<<<g, B, 0, stream>>>(y1, NN, H1D, sums, sumsq);
    }
    k_bn_finalize<<<cdiv(H1D, B), B, 0, stream>>>(sums, sumsq, g1, beta1, scale, shift, NN, H1D);
    k_bn_relu_f16<<<cdiv((long long)NN * H1D, B), B, 0, stream>>>(
        y1, scale, shift, h1h, (long long)NN * H1D, H1D);

    // ---- GEMM2: t2 = h1 @ W2 (N padded 300 -> 320) ----
    k_transpose_f16<<<cdiv((long long)H2P * H1D, B), B, 0, stream>>>(W2, w2t, H1D, H2D, H2P);
    {
        int nGrpN = H2P / 64;                          // 5 groups of 4 N-tiles
        int totalGrps = (NN / 16) * nGrpN;             // 3125 * 5
        k_wmma_gemm4<<<cdiv(totalGrps, B / 32), B, 0, stream>>>(
            h1h, w2t, t2, nullptr, H1D, nGrpN, H2D, H2D, totalGrps);
    }

    // ---- layer 2: propagate(t2) + b2 ----
    k_agg2_init<<<cdiv((long long)NN * H2D, B), B, 0, stream>>>(t2, dinv, b2, agg2);
    k_edge_scatter<<<cdiv((long long)NE * (H2D / 4), B), B, 0, stream>>>(
        erow, ecol, ew, dinv, t2, agg2, H2D, H2D / 4);

    // ---- BN2 + ReLU (in place, f32) ----
    k_zero_f32<<<cdiv(2 * H1D, B), B, 0, stream>>>(sums, 2 * H1D);
    {
        dim3 g(H2D, 32);
        k_bn_stats<<<g, B, 0, stream>>>(agg2, NN, H2D, sums, sumsq);
    }
    k_bn_finalize<<<cdiv(H2D, B), B, 0, stream>>>(sums, sumsq, g2, beta2, scale, shift, NN, H2D);
    k_bn_relu_f32<<<cdiv((long long)NN * H2D, B), B, 0, stream>>>(
        agg2, scale, shift, (long long)NN * H2D, H2D);

    // ---- head: relu(h2 @ Wl + bl) + log_softmax ----
    k_head<<<cdiv(NN, B), B, 0, stream>>>(agg2, Wl, bl, outp, NN);
}
